// RNN_75797582840813
// MI455X (gfx1250) — matrix operand order repaired and checked by reading, not compiled, not run on hardware
//
#include <hip/hip_runtime.h>

typedef __attribute__((ext_vector_type(16))) _Float16 v16h;
typedef __attribute__((ext_vector_type(8)))  _Float16 v8h;
typedef __attribute__((ext_vector_type(8)))  float    v8f;

namespace {
constexpr int H_   = 51;     // hidden size
constexpr int G3   = 153;    // 3*H
constexpr int NP   = 160;    // padded 3H (10 N-tiles of 16)
constexpr int KP   = 64;     // padded K  (2 K-tiles of 32)
constexpr int GST  = 165;    // gate-staging row stride in floats (odd -> no LDS bank conflicts)
constexpr int TOBS = 1000;
constexpr int TTOT = 2000;
constexpr int ROWS = 16;     // batch rows per wave (== WMMA M)

// LDS layout (bytes); all offsets 16B aligned
constexpr int OFF_W1H = 0;                       // w_hh1 f16 [160][64]
constexpr int OFF_W2I = OFF_W1H + NP * KP * 2;   // w_ih2 f16 [160][64]
constexpr int OFF_W2H = OFF_W2I + NP * KP * 2;   // w_hh2 f16 [160][64]
constexpr int OFF_H1  = OFF_W2H + NP * KP * 2;   // h1 f16 [16][64]
constexpr int OFF_H2  = OFF_H1  + ROWS * KP * 2; // h2 f16 [16][64]
constexpr int OFF_GI  = OFF_H2  + ROWS * KP * 2; // gate stage f32 [16][165]
constexpr int OFF_GH  = OFF_GI  + ROWS * GST * 4;// gate stage f32 [16][165]
constexpr int OFF_BI1 = OFF_GH  + ROWS * GST * 4;// b_ih1 f32 [160]
constexpr int OFF_BH1 = OFF_BI1 + NP * 4;
constexpr int OFF_BI2 = OFF_BH1 + NP * 4;
constexpr int OFF_BH2 = OFF_BI2 + NP * 4;
constexpr int OFF_WI1 = OFF_BH2 + NP * 4;        // w_ih1 (column vector) f32 [160]
constexpr int OFF_WLN = OFF_WI1 + NP * 4;        // w_lin f32 [64]
constexpr int OFF_XB  = OFF_WLN + KP * 4;        // per-row step input f32 [16]
constexpr int SMEM_BYTES = OFF_XB + ROWS * 4;    // ~90 KB (dynamic LDS; 320KB/WGP on CDNA5)
}  // namespace

__device__ __forceinline__ v8f wmma_f16f32(v16h a, v16h b, v8f c) {
  // v_wmma_f32_16x16x32_f16: D = A(16x32 f16) * B(32x16 f16) + C(16x16 f32)
  return __builtin_amdgcn_wmma_f32_16x16x32_f16(false, a, false, b, (short)0, c,
                                                false, false);
}

// A-matrix (16x32 f16) fragment from LDS h buffer [16][64].
// Per ISA 7.12.2: lanes 0-15 hold M=lane, K={kt*32+0..7, kt*32+16..23};
// lanes 16-31 hold M=lane-16, K={kt*32+8..15, kt*32+24..31}.
__device__ __forceinline__ v16h load_a_frag(const _Float16* hbase, int lane, int kt) {
  const int m  = lane & 15;
  const int k0 = kt * 32 + ((lane < 16) ? 0 : 8);
  const _Float16* p = hbase + m * KP + k0;
  v8h lo = *(const v8h*)(p);        // K = k0 .. k0+7
  v8h hi = *(const v8h*)(p + 16);   // K = k0+16 .. k0+23
  return __builtin_shufflevector(lo, hi, 0, 1, 2, 3, 4, 5, 6, 7,
                                 8, 9, 10, 11, 12, 13, 14, 15);
}

// B-matrix (32x16 f16) fragment from LDS weight buffer [160][64] (row = output n, col = k).
// Lanes 0-15: N=lane, K={kt*32+0..7, kt*32+16..23}; lanes 16-31: N=lane-16, K={kt*32+8..15, kt*32+24..31}.
__device__ __forceinline__ v16h load_b_frag(const _Float16* wbase, int lane, int nt, int kt) {
  const int n  = nt * 16 + (lane & 15);
  const int k0 = kt * 32 + ((lane < 16) ? 0 : 8);
  const _Float16* p = wbase + n * KP + k0;
  v8h lo = *(const v8h*)(p);
  v8h hi = *(const v8h*)(p + 16);
  return __builtin_shufflevector(lo, hi, 0, 1, 2, 3, 4, 5, 6, 7,
                                 8, 9, 10, 11, 12, 13, 14, 15);
}

// C/D layout: lane holds column nt*16+(lane&15); VGPR i holds row i (lanes 0-15) or i+8.
__device__ __forceinline__ void store_acc(float* g, int lane, int nt, v8f acc) {
  const int col = nt * 16 + (lane & 15);
  const int m0  = (lane < 16) ? 0 : 8;
#pragma unroll
  for (int i = 0; i < 8; ++i) g[(m0 + i) * GST + col] = acc[i];
}

__device__ __forceinline__ float sig_(float x) {
  return 1.0f / (1.0f + __expf(-x));
}
__device__ __forceinline__ float tanh_(float x) {
  x = fminf(fmaxf(x, -15.0f), 15.0f);
  const float e = __expf(-2.0f * x);
  return (1.0f - e) / (1.0f + e);
}

__global__ void __launch_bounds__(32)
gru_persist_kernel(const float* __restrict__ x,      // [2048,1000]
                   const float* __restrict__ wih1g,  // [153,1]
                   const float* __restrict__ whh1g,  // [153,51]
                   const float* __restrict__ bih1g,  // [153]
                   const float* __restrict__ bhh1g,  // [153]
                   const float* __restrict__ wih2g,  // [153,51]
                   const float* __restrict__ whh2g,  // [153,51]
                   const float* __restrict__ bih2g,  // [153]
                   const float* __restrict__ bhh2g,  // [153]
                   const float* __restrict__ wling,  // [51]
                   const float* __restrict__ bling,  // [1]
                   float* __restrict__ out)          // [2048,2000]
{
  extern __shared__ char smem[];
  _Float16* W1H = (_Float16*)(smem + OFF_W1H);
  _Float16* W2I = (_Float16*)(smem + OFF_W2I);
  _Float16* W2H = (_Float16*)(smem + OFF_W2H);
  _Float16* H1  = (_Float16*)(smem + OFF_H1);
  _Float16* H2  = (_Float16*)(smem + OFF_H2);
  float*    GI  = (float*)(smem + OFF_GI);
  float*    GH  = (float*)(smem + OFF_GH);
  float*    BI1 = (float*)(smem + OFF_BI1);
  float*    BH1 = (float*)(smem + OFF_BH1);
  float*    BI2 = (float*)(smem + OFF_BI2);
  float*    BH2 = (float*)(smem + OFF_BH2);
  float*    WI1 = (float*)(smem + OFF_WI1);
  float*    WLN = (float*)(smem + OFF_WLN);
  float*    XB  = (float*)(smem + OFF_XB);

  const int lane = threadIdx.x;          // 0..31 (single wave32 per block)
  const int rowb = blockIdx.x * ROWS;    // first batch row of this block

  // ---- one-time init: pack weights into WMMA-friendly f16 LDS layout ----
  {
    const float* src[3] = {whh1g, wih2g, whh2g};
    _Float16* dst[3]    = {W1H, W2I, W2H};
#pragma unroll
    for (int mi = 0; mi < 3; ++mi) {
      const float* s = src[mi];
      _Float16*    d = dst[mi];
      for (int idx = lane; idx < NP * KP; idx += 32) {
        const int n = idx >> 6, k = idx & 63;
        const float v = (n < G3 && k < H_) ? s[n * H_ + k] : 0.0f;
        d[idx] = (_Float16)v;
      }
    }
  }
  for (int c = lane; c < NP; c += 32) {
    BI1[c] = (c < G3) ? bih1g[c] : 0.0f;
    BH1[c] = (c < G3) ? bhh1g[c] : 0.0f;
    BI2[c] = (c < G3) ? bih2g[c] : 0.0f;
    BH2[c] = (c < G3) ? bhh2g[c] : 0.0f;
    WI1[c] = (c < G3) ? wih1g[c] : 0.0f;
  }
  for (int c = lane; c < KP; c += 32) WLN[c] = (c < H_) ? wling[c] : 0.0f;
  for (int idx = lane; idx < ROWS * KP; idx += 32) {
    H1[idx] = (_Float16)0.0f;
    H2[idx] = (_Float16)0.0f;
  }
  const float blin = bling[0];
  __syncthreads();

  // ---- 2000 sequential steps, everything resident ----
  for (int t = 0; t < TTOT; ++t) {
    // step input: observed x for t<1000, else previous prediction (already in XB)
    if (t < TOBS && lane < ROWS) {
      const float* xp = x + (size_t)(rowb + lane) * TOBS + t;
      XB[lane] = *xp;
      if (t + 8 < TOBS) __builtin_prefetch(xp + 8, 0, 0);
    }

    // ---- cell1: gh1 = h1 @ w_hh1^T  (10 N-tiles x 2 K-tiles) ----
    // B fragments software-pipelined one N-tile ahead so ds_load latency
    // overlaps the previous tile's WMMAs (enables partial s_wait_dscnt).
    {
      const v16h a0 = load_a_frag(H1, lane, 0);
      const v16h a1 = load_a_frag(H1, lane, 1);
      v16h nb0 = load_b_frag(W1H, lane, 0, 0);
      v16h nb1 = load_b_frag(W1H, lane, 0, 1);
#pragma unroll
      for (int nt = 0; nt < 10; ++nt) {
        const v16h cb0 = nb0, cb1 = nb1;
        if (nt < 9) {
          nb0 = load_b_frag(W1H, lane, nt + 1, 0);
          nb1 = load_b_frag(W1H, lane, nt + 1, 1);
        }
        v8f acc = {};
        acc = wmma_f16f32(a0, cb0, acc);
        acc = wmma_f16f32(a1, cb1, acc);
        store_acc(GH, lane, nt, acc);
      }
    }
    __syncthreads();

    // ---- cell1 elementwise: h1' (gi1 is scalar-input, computed inline) ----
#pragma unroll 1
    for (int it = 0; it < 26; ++it) {
      const int j = (lane >> 4) + 2 * it;
      if (j < H_) {
        const int m = lane & 15;
        const float xv = XB[m];
        const float* gh = GH + m * GST;
        const float r = sig_(xv * WI1[j]        + BI1[j]        + gh[j]        + BH1[j]);
        const float z = sig_(xv * WI1[j + H_]   + BI1[j + H_]   + gh[j + H_]   + BH1[j + H_]);
        const float n = tanh_(xv * WI1[j + 2*H_] + BI1[j + 2*H_] +
                              r * (gh[j + 2*H_] + BH1[j + 2*H_]));
        const float hold = (float)H1[m * KP + j];
        H1[m * KP + j] = (_Float16)((1.0f - z) * n + z * hold);
      }
    }
    __syncthreads();

    // ---- cell2: gi2 = h1 @ w_ih2^T ; gh2 = h2 @ w_hh2^T ----
    // N-tiles 0..5 (cols < 96, pure r/z): chain both matmuls into ONE accumulator.
    // N-tiles 6..9 (cols >= 96, includes n-gate): keep gi and gh separate.
    // All four B fragments double-buffered one N-tile ahead.
    {
      const v16h p0 = load_a_frag(H1, lane, 0);
      const v16h p1 = load_a_frag(H1, lane, 1);
      const v16h q0 = load_a_frag(H2, lane, 0);
      const v16h q1 = load_a_frag(H2, lane, 1);
      v16h ni0 = load_b_frag(W2I, lane, 0, 0);
      v16h ni1 = load_b_frag(W2I, lane, 0, 1);
      v16h nh0 = load_b_frag(W2H, lane, 0, 0);
      v16h nh1 = load_b_frag(W2H, lane, 0, 1);
#pragma unroll
      for (int nt = 0; nt < 10; ++nt) {
        const v16h ci0 = ni0, ci1 = ni1, ch0 = nh0, ch1 = nh1;
        if (nt < 9) {
          ni0 = load_b_frag(W2I, lane, nt + 1, 0);
          ni1 = load_b_frag(W2I, lane, nt + 1, 1);
          nh0 = load_b_frag(W2H, lane, nt + 1, 0);
          nh1 = load_b_frag(W2H, lane, nt + 1, 1);
        }
        if (nt < 6) {
          v8f acc = {};
          acc = wmma_f16f32(p0, ci0, acc);
          acc = wmma_f16f32(p1, ci1, acc);
          acc = wmma_f16f32(q0, ch0, acc);
          acc = wmma_f16f32(q1, ch1, acc);
          store_acc(GI, lane, nt, acc);
        } else {
          v8f ai = {};
          ai = wmma_f16f32(p0, ci0, ai);
          ai = wmma_f16f32(p1, ci1, ai);
          store_acc(GI, lane, nt, ai);
          v8f ah = {};
          ah = wmma_f16f32(q0, ch0, ah);
          ah = wmma_f16f32(q1, ch1, ah);
          store_acc(GH, lane, nt, ah);
        }
      }
    }
    __syncthreads();

    // ---- cell2 elementwise: h2' ----
#pragma unroll 1
    for (int it = 0; it < 26; ++it) {
      const int j = (lane >> 4) + 2 * it;
      if (j < H_) {
        const int m = lane & 15;
        const float* gi = GI + m * GST;
        const float* gh = GH + m * GST;
        const float r = sig_(gi[j] + BI2[j] + BH2[j]);  // cols<96: gi holds gi+gh
        const int cz = j + H_;
        const float sz = (cz < 96) ? (gi[cz] + BI2[cz] + BH2[cz])
                                   : (gi[cz] + gh[cz] + BI2[cz] + BH2[cz]);
        const float z = sig_(sz);
        const int cn = j + 2 * H_;  // >= 102: separate region
        const float n = tanh_(gi[cn] + BI2[cn] + r * (gh[cn] + BH2[cn]));
        const float hold = (float)H2[m * KP + j];
        H2[m * KP + j] = (_Float16)((1.0f - z) * n + z * hold);
      }
    }
    __syncthreads();

    // ---- output head: out = h2 @ w_lin^T + b_lin ; feed back for future steps ----
    // Dot product split across the two half-waves, combined with a lane shuffle.
    {
      const int m = lane & 15;
      const int j0 = (lane >> 4) * 26;  // lanes 0-15: j=0..25, lanes 16-31: j=26..50
      const _Float16* hr = H2 + m * KP;
      float part = 0.0f;
#pragma unroll
      for (int k = 0; k < 26; ++k) {
        const int j = j0 + k;
        if (j < H_) part += (float)hr[j] * WLN[j];
      }
      part += __shfl_xor(part, 16, 32);
      if (lane < ROWS) {
        const float acc = part + blin;
        out[(size_t)(rowb + lane) * TTOT + t] = acc;
        XB[lane] = acc;  // only consumed when next step is autoregressive
      }
    }
    __syncthreads();
  }
}

extern "C" void kernel_launch(void* const* d_in, const int* in_sizes, int n_in,
                              void* d_out, int out_size, void* d_ws, size_t ws_size,
                              hipStream_t stream) {
  (void)in_sizes; (void)n_in; (void)d_ws; (void)ws_size; (void)out_size;
  const float* x     = (const float*)d_in[0];
  const float* wih1  = (const float*)d_in[1];
  const float* whh1  = (const float*)d_in[2];
  const float* bih1  = (const float*)d_in[3];
  const float* bhh1  = (const float*)d_in[4];
  const float* wih2  = (const float*)d_in[5];
  const float* whh2  = (const float*)d_in[6];
  const float* bih2  = (const float*)d_in[7];
  const float* bhh2  = (const float*)d_in[8];
  const float* wlin  = (const float*)d_in[9];
  const float* blin  = (const float*)d_in[10];
  float* out = (float*)d_out;

  const int B = 2048;
  dim3 grid(B / ROWS);   // 128 independent blocks, one wave32 each, 16 batch rows/wave
  dim3 block(32);
  gru_persist_kernel<<<grid, block, SMEM_BYTES, stream>>>(
      x, wih1, whh1, bih1, bhh1, wih2, whh2, bih2, bhh2, wlin, blin, out);
}
